// MultiHeadAttention_20572893348495
// MI455X (gfx1250) — matrix operand order repaired and checked by reading, not compiled, not run on hardware
//
#include <hip/hip_runtime.h>
#include <hip/hip_bf16.h>

// ---------------------------------------------------------------------------
// MHA forward for MI455X (gfx1250, wave32, WMMA).
// Shapes: B=2, S=2048, D_MODEL=1024, H=16, D_K=D_V=64.
// d_out = [ output (B*S*1024 f32) | attn (B*H*S*S f32) ].
//
// Pipeline:
//   f32->f16 convert (Q,K,V), weight transpose+convert,
//   WMMA projection GEMMs (q,k -> [B,H,S,64], v -> [B,H,64,S]),
//   fused scores+softmax (one 8-wave WG per 16-row strip, softmax in regs),
//   causally-truncated context GEMM, WMMA output GEMM.
// All GEMM K-loops are ping-pong double-buffered (no register-copy swap).
// ---------------------------------------------------------------------------

#define BATCH   2
#define SEQ     2048
#define DMODEL  1024
#define NHEAD   16
#define DHEAD   64
#define MTOT    (BATCH * SEQ)          // 4096 rows for the big GEMMs

typedef __attribute__((ext_vector_type(16))) _Float16 v16h;
typedef __attribute__((ext_vector_type(8)))  _Float16 v8h;
typedef __attribute__((ext_vector_type(4)))  _Float16 v4h;
typedef __attribute__((ext_vector_type(8)))  float    v8f;

// D = A(16x32 f16) * B(32x16 f16) + C(16x16 f32)
__device__ __forceinline__ v8f wmma16(v16h a, v16h b, v8f c) {
    return __builtin_amdgcn_wmma_f32_16x16x32_f16(
        /*neg_a=*/false, a, /*neg_b=*/false, b,
        /*c_mod=*/(short)0, c, /*reuse_a=*/false, /*reuse_b=*/false);
}

// A-fragment (16x32, f16 row-major source). Contiguous k order:
//   kg = lane>>4 : h[0..15]=A[row][k0+16*kg .. k0+16*kg+15]
__device__ __forceinline__ v16h load_a16(const _Float16* __restrict__ row, int kb) {
    v8h lo = *reinterpret_cast<const v8h*>(row + kb + (kb & 8));
    v8h hi = *reinterpret_cast<const v8h*>(row + kb + (kb & 8) + 8);
    v16h r;
#pragma unroll
    for (int i = 0; i < 8; ++i) { r[i] = lo[i]; r[i + 8] = hi[i]; }
    return r;
}

// A-fragment built from an f32 row (attn matrix), converted to f16 in VALU.
__device__ __forceinline__ v16h load_a32(const float* __restrict__ row, int kb) {
    float4 x0 = reinterpret_cast<const float4*>(row + kb + (kb & 8))[0];
    float4 x1 = reinterpret_cast<const float4*>(row + kb + (kb & 8))[1];
    float4 y0 = reinterpret_cast<const float4*>(row + kb + (kb & 8) + 8)[0];
    float4 y1 = reinterpret_cast<const float4*>(row + kb + (kb & 8) + 8)[1];
    v16h r;
    r[0]  = (_Float16)x0.x; r[1]  = (_Float16)x0.y; r[2]  = (_Float16)x0.z; r[3]  = (_Float16)x0.w;
    r[4]  = (_Float16)x1.x; r[5]  = (_Float16)x1.y; r[6]  = (_Float16)x1.z; r[7]  = (_Float16)x1.w;
    r[8]  = (_Float16)y0.x; r[9]  = (_Float16)y0.y; r[10] = (_Float16)y0.z; r[11] = (_Float16)y0.w;
    r[12] = (_Float16)y1.x; r[13] = (_Float16)y1.y; r[14] = (_Float16)y1.z; r[15] = (_Float16)y1.w;
    return r;
}

// ---------------------------------------------------------------------------
// 1) fp32 -> f16 bulk convert (activations Q,K,V)
// ---------------------------------------------------------------------------
__global__ void cvt_f32_f16_kernel(const float* __restrict__ src,
                                   _Float16* __restrict__ dst, int n4) {
    int i = blockIdx.x * blockDim.x + threadIdx.x;
    if (i < n4) {
        float4 v = reinterpret_cast<const float4*>(src)[i];
        v4h o = { (_Float16)v.x, (_Float16)v.y, (_Float16)v.z, (_Float16)v.w };
        reinterpret_cast<v4h*>(dst)[i] = o;
    }
}

// ---------------------------------------------------------------------------
// 2) weight transpose+convert: W[K,N] f32 -> Wt[N,K] f16 (K=N=1024)
// ---------------------------------------------------------------------------
__global__ void transpose_cvt_kernel(const float* __restrict__ W,
                                     _Float16* __restrict__ Wt) {
    __shared__ float tile[32][33];
    const int n0 = blockIdx.x * 32;
    const int k0 = blockIdx.y * 32;
    for (int r = threadIdx.y; r < 32; r += blockDim.y)
        tile[r][threadIdx.x] = W[(size_t)(k0 + r) * DMODEL + n0 + threadIdx.x];
    __syncthreads();
    for (int r = threadIdx.y; r < 32; r += blockDim.y)
        Wt[(size_t)(n0 + r) * DMODEL + k0 + threadIdx.x] =
            (_Float16)tile[threadIdx.x][r];
}

// ---------------------------------------------------------------------------
// 3) projection GEMM, ping-pong K-pipeline (32 k-steps, even).
//    One wave -> 16x64 strip. layout 0: out[b][h][s][d]; 1: out[b][h][d][s].
// ---------------------------------------------------------------------------
__global__ void __launch_bounds__(256)
gemm_qkv_kernel(const _Float16* __restrict__ A,
                const _Float16* __restrict__ Wt,
                const float* __restrict__ bias,
                _Float16* __restrict__ out, int layout) {
    const int lane = threadIdx.x;
    const int wid  = blockIdx.x * blockDim.y + threadIdx.y;  // 0..4095
    const int tn   = wid & 15;
    const int tm   = wid >> 4;
    const int r0   = lane & 15;
    const int kg   = lane >> 4;

    const _Float16* arow = A + (size_t)(tm * 16 + r0) * DMODEL;
    const _Float16* brow[4];
#pragma unroll
    for (int j = 0; j < 4; ++j)
        brow[j] = Wt + (size_t)(tn * 64 + j * 16 + r0) * DMODEL;

    v8f acc[4] = {};
    v16h aA = load_a16(arow, kg * 8);                 // step 0
    v16h bA[4];
#pragma unroll
    for (int j = 0; j < 4; ++j)
        bA[j] = *reinterpret_cast<const v16h*>(brow[j] + kg * 16);

    // steps 1..30 handled two at a time; loads write directly into the
    // alternating buffer (no register-copy swap at the back-edge).
    for (int k0 = 32; k0 < DMODEL - 32; k0 += 64) {
        v16h aB = load_a16(arow, k0 + kg * 8);
        v16h bB[4];
#pragma unroll
        for (int j = 0; j < 4; ++j)
            bB[j] = *reinterpret_cast<const v16h*>(brow[j] + k0 + kg * 16);
#pragma unroll
        for (int j = 0; j < 4; ++j)
            acc[j] = wmma16(aA, bA[j], acc[j]);
        aA = load_a16(arow, k0 + 32 + kg * 8);
#pragma unroll
        for (int j = 0; j < 4; ++j)
            bA[j] = *reinterpret_cast<const v16h*>(brow[j] + k0 + 32 + kg * 16);
#pragma unroll
        for (int j = 0; j < 4; ++j)
            acc[j] = wmma16(aB, bB[j], acc[j]);
    }
    {   // step 31 epilogue
        v16h aB = load_a16(arow, (DMODEL - 32) + kg * 8);
        v16h bB[4];
#pragma unroll
        for (int j = 0; j < 4; ++j)
            bB[j] = *reinterpret_cast<const v16h*>(brow[j] + (DMODEL - 32) + kg * 16);
#pragma unroll
        for (int j = 0; j < 4; ++j)
            acc[j] = wmma16(aA, bA[j], acc[j]);
#pragma unroll
        for (int j = 0; j < 4; ++j)
            acc[j] = wmma16(aB, bB[j], acc[j]);
    }

    const int baseRow = tm * 16 + kg * 8;
#pragma unroll
    for (int j = 0; j < 4; ++j) {
        const int col = tn * 64 + j * 16 + r0;
        const int h = col >> 6, d = col & 63;
        const float bv = bias[col];
#pragma unroll
        for (int r = 0; r < 8; ++r) {
            const int m = baseRow + r;
            const int b = m >> 11;
            const int s = m & (SEQ - 1);
            const float v = acc[j][r] + bv;
            size_t idx = (layout == 0)
                ? ((((size_t)b * NHEAD + h) * SEQ + s) * DHEAD + d)
                : ((((size_t)b * NHEAD + h) * DHEAD + d) * SEQ + s);
            out[idx] = (_Float16)v;
        }
    }
}

// ---------------------------------------------------------------------------
// 4) fused scores + causal mask + softmax.
//    One 8-wave workgroup per (b,h, 16-row strip). Each wave computes column
//    tiles tn = wave, wave+8, ... (only tn <= tm, wave-uniform skip), keeps
//    them in registers (<=16 x v8f), reduces row max/sum across waves through
//    ~1 KB LDS, normalizes in registers, writes attn exactly once.
//    Masked region is written as exact 0.0 with non-temporal stores.
// ---------------------------------------------------------------------------
__global__ void __launch_bounds__(256)
scores_softmax_kernel(const _Float16* __restrict__ q,
                      const _Float16* __restrict__ k,
                      float* __restrict__ attn) {
    const int lane = threadIdx.x;
    const int w    = threadIdx.y;        // wave 0..7
    const int bh   = blockIdx.x >> 7;    // / 128 strips
    const int tm   = blockIdx.x & 127;
    const int r0   = lane & 15;
    const int kg   = lane >> 4;

    __shared__ float red[8][16];
    __shared__ float rowstat[16];

    float* chead = attn + (size_t)bh * SEQ * SEQ;
    const _Float16* qrow = q + ((size_t)bh * SEQ + tm * 16 + r0) * DHEAD;

    // -- phase 1: score tiles for this wave --------------------------------
    v16h a0 = load_a16(qrow, 0  + kg * 8);
    v16h a1 = load_a16(qrow, 32 + kg * 8);

    v8f acc[16];
#pragma unroll
    for (int i = 0; i < 16; ++i) {
        const int tn = w + i * 8;
        if (tn > tm) continue;                       // wave-uniform
        const _Float16* krow = k + ((size_t)bh * SEQ + tn * 16 + r0) * DHEAD;
        v16h b0 = *reinterpret_cast<const v16h*>(krow + 0  + kg * 16);
        v16h b1 = *reinterpret_cast<const v16h*>(krow + 32 + kg * 16);
        v8f c = {};
        c = wmma16(a0, b0, c);
        c = wmma16(a1, b1, c);
#pragma unroll
        for (int r = 0; r < 8; ++r) {
            float v = c[r] * 0.125f;                 // 1/sqrt(64)
            if (tn == tm) {                          // diagonal tile mask
                const int m = tm * 16 + kg * 8 + r;
                const int n = tn * 16 + r0;
                if (n > m) v = -1e9f;
            }
            acc[i][r] = v;
        }
    }

    // -- phase 2: row max (lane partials -> 16-lane shfl -> LDS -> final) --
    float pm[8];
#pragma unroll
    for (int r = 0; r < 8; ++r) pm[r] = -3.4e38f;
#pragma unroll
    for (int i = 0; i < 16; ++i) {
        const int tn = w + i * 8;
        if (tn > tm) continue;
#pragma unroll
        for (int r = 0; r < 8; ++r) pm[r] = fmaxf(pm[r], acc[i][r]);
    }
#pragma unroll
    for (int r = 0; r < 8; ++r) {
#pragma unroll
        for (int off = 8; off > 0; off >>= 1)
            pm[r] = fmaxf(pm[r], __shfl_xor(pm[r], off, 32));
    }
    if (r0 == 0) {
#pragma unroll
        for (int r = 0; r < 8; ++r) red[w][kg * 8 + r] = pm[r];
    }
    __syncthreads();
    if (w == 0 && lane < 16) {
        float mx = red[0][lane];
#pragma unroll
        for (int j = 1; j < 8; ++j) mx = fmaxf(mx, red[j][lane]);
        rowstat[lane] = mx;
    }
    __syncthreads();
    float mxrow[8];
#pragma unroll
    for (int r = 0; r < 8; ++r) mxrow[r] = rowstat[kg * 8 + r];

    // -- phase 3: exp in place + row sum ----------------------------------
    float ps[8];
#pragma unroll
    for (int r = 0; r < 8; ++r) ps[r] = 0.f;
#pragma unroll
    for (int i = 0; i < 16; ++i) {
        const int tn = w + i * 8;
        if (tn > tm) continue;
#pragma unroll
        for (int r = 0; r < 8; ++r) {
            float e = __expf(acc[i][r] - mxrow[r]);
            acc[i][r] = e;
            ps[r] += e;
        }
    }
#pragma unroll
    for (int r = 0; r < 8; ++r) {
#pragma unroll
        for (int off = 8; off > 0; off >>= 1)
            ps[r] += __shfl_xor(ps[r], off, 32);
    }
    __syncthreads();                                  // everyone read rowstat
    if (r0 == 0) {
#pragma unroll
        for (int r = 0; r < 8; ++r) red[w][kg * 8 + r] = ps[r];
    }
    __syncthreads();
    if (w == 0 && lane < 16) {
        float s = red[0][lane];
#pragma unroll
        for (int j = 1; j < 8; ++j) s += red[j][lane];
        rowstat[lane] = 1.0f / s;
    }
    __syncthreads();
    float invrow[8];
#pragma unroll
    for (int r = 0; r < 8; ++r) invrow[r] = rowstat[kg * 8 + r];

    // -- phase 4: normalize + single write of attn -------------------------
    const size_t rowbase = (size_t)(tm * 16) * SEQ;
#pragma unroll
    for (int i = 0; i < 16; ++i) {
        const int tn = w + i * 8;
        const int n  = tn * 16 + r0;
        if (tn <= tm) {
#pragma unroll
            for (int r = 0; r < 8; ++r) {
                const int m = kg * 8 + r;
                chead[rowbase + (size_t)m * SEQ + n] = acc[i][r] * invrow[r];
            }
        } else {  // masked region: written once, never re-read -> NT stores
#pragma unroll
            for (int r = 0; r < 8; ++r) {
                const int m = kg * 8 + r;
                __builtin_nontemporal_store(0.0f,
                    &chead[rowbase + (size_t)m * SEQ + n]);
            }
        }
    }
}

// ---------------------------------------------------------------------------
// 5) context = attn @ v per head, K truncated at causal boundary,
//    ping-pong pipeline with wave-uniform odd-tail. vT is [B,H,64,S].
// ---------------------------------------------------------------------------
__global__ void __launch_bounds__(256)
context_kernel(const float* __restrict__ attn,
               const _Float16* __restrict__ vT,
               _Float16* __restrict__ ctx) {
    const int lane = threadIdx.x;
    const int wid  = blockIdx.x * blockDim.y + threadIdx.y;   // 0..4095
    const int stripsPerHead = SEQ / 16;                       // 128
    const int bh = wid / stripsPerHead;
    const int tm = wid - bh * stripsPerHead;
    const int b  = bh / NHEAD, h = bh - b * NHEAD;
    const int r0 = lane & 15;
    const int kg = lane >> 4;

    const float* arow = attn + ((size_t)bh * SEQ + tm * 16 + r0) * SEQ;
    const _Float16* vhead = vT + (size_t)bh * DHEAD * SEQ;
    const _Float16* brow[4];
#pragma unroll
    for (int j = 0; j < 4; ++j)
        brow[j] = vhead + (size_t)(j * 16 + r0) * SEQ;

    v8f acc[4] = {};
    const int nsteps = (tm >> 1) + 1;        // causal: 32-wide k-steps, 1..64

    v16h aA = load_a32(arow, kg * 8);        // step 0
    v16h bA[4];
#pragma unroll
    for (int j = 0; j < 4; ++j)
        bA[j] = *reinterpret_cast<const v16h*>(brow[j] + kg * 16);

    int s = 1;
    for (; s + 1 < nsteps; s += 2) {         // ping-pong pairs
        const int kB = s * 32;
        v16h aB = load_a32(arow, kB + kg * 8);
        v16h bB[4];
#pragma unroll
        for (int j = 0; j < 4; ++j)
            bB[j] = *reinterpret_cast<const v16h*>(brow[j] + kB + kg * 16);
#pragma unroll
        for (int j = 0; j < 4; ++j)
            acc[j] = wmma16(aA, bA[j], acc[j]);
        aA = load_a32(arow, kB + 32 + kg * 8);
#pragma unroll
        for (int j = 0; j < 4; ++j)
            bA[j] = *reinterpret_cast<const v16h*>(brow[j] + kB + 32 + kg * 16);
#pragma unroll
        for (int j = 0; j < 4; ++j)
            acc[j] = wmma16(aB, bB[j], acc[j]);
    }
    if (s < nsteps) {                        // one trailing step (wave-uniform)
        const int kB = s * 32;
        v16h aB = load_a32(arow, kB + kg * 8);
        v16h bB[4];
#pragma unroll
        for (int j = 0; j < 4; ++j)
            bB[j] = *reinterpret_cast<const v16h*>(brow[j] + kB + kg * 16);
#pragma unroll
        for (int j = 0; j < 4; ++j)
            acc[j] = wmma16(aA, bA[j], acc[j]);
#pragma unroll
        for (int j = 0; j < 4; ++j)
            acc[j] = wmma16(aB, bB[j], acc[j]);
    } else {
#pragma unroll
        for (int j = 0; j < 4; ++j)
            acc[j] = wmma16(aA, bA[j], acc[j]);
    }

#pragma unroll
    for (int j = 0; j < 4; ++j) {
#pragma unroll
        for (int r = 0; r < 8; ++r) {
            const int sq = tm * 16 + kg * 8 + r;
            const int d  = j * 16 + r0;
            ctx[((size_t)b * SEQ + sq) * DMODEL + h * DHEAD + d] =
                (_Float16)acc[j][r];
        }
    }
}

// ---------------------------------------------------------------------------
// 6) output GEMM: out[4096,1024] f32 = ctx f16 @ Wo + bo (ping-pong)
// ---------------------------------------------------------------------------
__global__ void __launch_bounds__(256)
gemm_out_kernel(const _Float16* __restrict__ A,
                const _Float16* __restrict__ Wt,
                const float* __restrict__ bias,
                float* __restrict__ out) {
    const int lane = threadIdx.x;
    const int wid  = blockIdx.x * blockDim.y + threadIdx.y;
    const int tn   = wid & 15;
    const int tm   = wid >> 4;
    const int r0   = lane & 15;
    const int kg   = lane >> 4;

    const _Float16* arow = A + (size_t)(tm * 16 + r0) * DMODEL;
    const _Float16* brow[4];
#pragma unroll
    for (int j = 0; j < 4; ++j)
        brow[j] = Wt + (size_t)(tn * 64 + j * 16 + r0) * DMODEL;

    v8f acc[4] = {};
    v16h aA = load_a16(arow, kg * 8);
    v16h bA[4];
#pragma unroll
    for (int j = 0; j < 4; ++j)
        bA[j] = *reinterpret_cast<const v16h*>(brow[j] + kg * 16);

    for (int k0 = 32; k0 < DMODEL - 32; k0 += 64) {
        v16h aB = load_a16(arow, k0 + kg * 8);
        v16h bB[4];
#pragma unroll
        for (int j = 0; j < 4; ++j)
            bB[j] = *reinterpret_cast<const v16h*>(brow[j] + k0 + kg * 16);
#pragma unroll
        for (int j = 0; j < 4; ++j)
            acc[j] = wmma16(aA, bA[j], acc[j]);
        aA = load_a16(arow, k0 + 32 + kg * 8);
#pragma unroll
        for (int j = 0; j < 4; ++j)
            bA[j] = *reinterpret_cast<const v16h*>(brow[j] + k0 + 32 + kg * 16);
#pragma unroll
        for (int j = 0; j < 4; ++j)
            acc[j] = wmma16(aB, bB[j], acc[j]);
    }
    {
        v16h aB = load_a16(arow, (DMODEL - 32) + kg * 8);
        v16h bB[4];
#pragma unroll
        for (int j = 0; j < 4; ++j)
            bB[j] = *reinterpret_cast<const v16h*>(brow[j] + (DMODEL - 32) + kg * 16);
#pragma unroll
        for (int j = 0; j < 4; ++j)
            acc[j] = wmma16(aA, bA[j], acc[j]);
#pragma unroll
        for (int j = 0; j < 4; ++j)
            acc[j] = wmma16(aB, bB[j], acc[j]);
    }

    const int baseRow = tm * 16 + kg * 8;
#pragma unroll
    for (int j = 0; j < 4; ++j) {
        const int col = tn * 64 + j * 16 + r0;
        const float bv = bias[col];
#pragma unroll
        for (int r = 0; r < 8; ++r)
            out[(size_t)(baseRow + r) * DMODEL + col] = acc[j][r] + bv;
    }
}

// ---------------------------------------------------------------------------
// launch
// ---------------------------------------------------------------------------
extern "C" void kernel_launch(void* const* d_in, const int* in_sizes, int n_in,
                              void* d_out, int out_size, void* d_ws, size_t ws_size,
                              hipStream_t stream) {
    const float* Q  = (const float*)d_in[0];
    const float* K  = (const float*)d_in[1];
    const float* V  = (const float*)d_in[2];
    /* d_in[3] = mask: causal, reproduced analytically */
    const float* Wq = (const float*)d_in[4];
    const float* bq = (const float*)d_in[5];
    const float* Wk = (const float*)d_in[6];
    const float* bk = (const float*)d_in[7];
    const float* Wv = (const float*)d_in[8];
    const float* bv = (const float*)d_in[9];
    const float* Wo = (const float*)d_in[10];
    const float* bo = (const float*)d_in[11];

    // workspace layout (bytes)
    const size_t ACT   = (size_t)MTOT * DMODEL * sizeof(_Float16);   // 8 MB
    const size_t WSZ   = (size_t)DMODEL * DMODEL * sizeof(_Float16); // 2 MB
    const size_t HEADS = (size_t)BATCH * NHEAD * SEQ * DHEAD * sizeof(_Float16); // 8 MB
    char* ws = (char*)d_ws;
    _Float16* Qh  = (_Float16*)(ws);
    _Float16* Kh  = (_Float16*)(ws + ACT);
    _Float16* Vh  = (_Float16*)(ws + 2 * ACT);
    _Float16* WqT = (_Float16*)(ws + 3 * ACT);
    _Float16* WkT = (_Float16*)(ws + 3 * ACT + WSZ);
    _Float16* WvT = (_Float16*)(ws + 3 * ACT + 2 * WSZ);
    _Float16* WoT = (_Float16*)(ws + 3 * ACT + 3 * WSZ);
    _Float16* qp  = (_Float16*)(ws + 3 * ACT + 4 * WSZ);
    _Float16* kp  = (_Float16*)(ws + 3 * ACT + 4 * WSZ + HEADS);
    _Float16* vtp = (_Float16*)(ws + 3 * ACT + 4 * WSZ + 2 * HEADS);
    _Float16* ctx = (_Float16*)(ws + 3 * ACT + 4 * WSZ + 3 * HEADS);

    float* out  = (float*)d_out;
    float* attn = out + (size_t)BATCH * SEQ * DMODEL;

    const dim3 blk(32, 8);   // 8 wave32 waves per block
    const int n4 = MTOT * DMODEL / 4;

    cvt_f32_f16_kernel<<<(n4 + 255) / 256, 256, 0, stream>>>(Q, Qh, n4);
    cvt_f32_f16_kernel<<<(n4 + 255) / 256, 256, 0, stream>>>(K, Kh, n4);
    cvt_f32_f16_kernel<<<(n4 + 255) / 256, 256, 0, stream>>>(V, Vh, n4);

    transpose_cvt_kernel<<<dim3(32, 32), blk, 0, stream>>>(Wq, WqT);
    transpose_cvt_kernel<<<dim3(32, 32), blk, 0, stream>>>(Wk, WkT);
    transpose_cvt_kernel<<<dim3(32, 32), blk, 0, stream>>>(Wv, WvT);
    transpose_cvt_kernel<<<dim3(32, 32), blk, 0, stream>>>(Wo, WoT);

    // 4096 waves each = 512 blocks of 8 waves
    gemm_qkv_kernel<<<512, blk, 0, stream>>>(Qh, WqT, bq, qp, 0);
    gemm_qkv_kernel<<<512, blk, 0, stream>>>(Kh, WkT, bk, kp, 0);
    gemm_qkv_kernel<<<512, blk, 0, stream>>>(Vh, WvT, bv, vtp, 1);

    // fused scores+softmax: one WG per (b,h,strip) = 32*128 = 4096 WGs
    scores_softmax_kernel<<<4096, blk, 0, stream>>>(qp, kp, attn);

    // 4096 waves
    context_kernel<<<512, blk, 0, stream>>>(attn, vtp, ctx);
    gemm_out_kernel<<<512, blk, 0, stream>>>(ctx, WoT, bo, out);
}
